// BernoulliMultiHeadAttention_30365418783474
// MI455X (gfx1250) — matrix operand order repaired and checked by reading, not compiled, not run on hardware
//
#include <hip/hip_runtime.h>

// ---------------------------------------------------------------------------
// BernoulliMultiHeadAttention forward for MI455X (gfx1250, wave32, WMMA).
// B=2, S=2048, E=1024, H=16, D=64.
// fp32->bf16 cvt -> QKV GEMM (WMMA bf16, V stored transposed [B,H,D,S]) ->
// fused sigmoid/Bernoulli attention (WMMA bf16, LDS sample staging) ->
// output GEMM (WMMA bf16). Software-pipelined fragment loads throughout.
// ---------------------------------------------------------------------------

typedef unsigned short u16;
typedef unsigned int   u32;
typedef __attribute__((ext_vector_type(16))) __bf16 v16bf;
typedef __attribute__((ext_vector_type(8)))  float  v8f;
typedef __attribute__((ext_vector_type(4)))  float  vf4;   // trivial 16B vector
typedef __attribute__((ext_vector_type(2)))  u32    vu2;

#define BATCH 2
#define S_LEN 2048
#define EMB   1024
#define NH    16
#define HD    64

union FragBF {
  v16bf bf;      // 16 bf16 = 8 VGPRs (A or B operand of 16x16x32 bf16 WMMA)
  u16   us[16];
  u32   ui[8];
  vf4   f4[2];
};

__device__ __forceinline__ u16 f2bf(float f) {           // RNE float->bf16
  u32 u = __float_as_uint(f);
  return (u16)((u + 0x7FFFu + ((u >> 16) & 1u)) >> 16);
}

__device__ __forceinline__ u32 pcg(u32 x) {              // counter-based hash
  x ^= x >> 16; x *= 0x7feb352du;
  x ^= x >> 15; x *= 0x846ca68bu;
  x ^= x >> 16;
  return x;
}

__device__ __forceinline__ v8f zero_v8f() {
  v8f z = {0.f, 0.f, 0.f, 0.f, 0.f, 0.f, 0.f, 0.f};
  return z;
}

__device__ __forceinline__ void ld_frag(FragBF& f, const u16* p0, const u16* p1) {
  f.f4[0] = *reinterpret_cast<const vf4*>(p0);
  f.f4[1] = *reinterpret_cast<const vf4*>(p1);
}

// --------------------------- fp32 -> bf16 convert ---------------------------
__global__ void cvt_bf16_kernel(const float* __restrict__ src,
                                u16* __restrict__ dst, int n4) {
  int i = blockIdx.x * blockDim.x + threadIdx.x;
  if (i >= n4) return;
  vf4 f = reinterpret_cast<const vf4*>(src)[i];
  vu2 o;
  o.x = (u32)f2bf(f.x) | ((u32)f2bf(f.y) << 16);
  o.y = (u32)f2bf(f.z) | ((u32)f2bf(f.w) << 16);
  reinterpret_cast<vu2*>(dst)[i] = o;
}

// ------------------------------- QKV GEMM ----------------------------------
// C[m,n] = sum_k X[m,k]*W[n,k] + bias[n]; one wave -> 16x64 tile.
// q,k stored [B,H,S,D] bf16; v stored transposed [B,H,D,S] bf16.
__global__ __launch_bounds__(128) void qkv_gemm_kernel(
    const u16* __restrict__ xb, const u16* __restrict__ wb,
    const float* __restrict__ bias,
    u16* __restrict__ qo, u16* __restrict__ ko, u16* __restrict__ vo) {
  const int lane = threadIdx.x & 31;
  const int wave = threadIdx.x >> 5;
  const int wid  = blockIdx.x * 4 + wave;          // 12288 waves
  const int nt   = wid % 48;                       // 48 tiles of N=64
  const int mt   = wid / 48;                       // 256 tiles of M=16
  const int l16  = lane & 15;
  const int hi8  = (lane >> 4) * 8;                // C/D-layout lane-half row shift
  const int hi16 = hi8 * 2;                        // B-layout lane-half shift

  const u16* arow = xb + (u32)(mt * 16 + l16) * EMB + hi16;
  const u16* wrow0 = wb + (u32)(nt * 64 + l16) * EMB + hi16;
  v8f acc[4] = {zero_v8f(), zero_v8f(), zero_v8f(), zero_v8f()};

  for (int kb = 0; kb < EMB; kb += 64) {
    FragBF a0, a1, b0[4], b1[4];                   // distinct regs -> pipelined
    ld_frag(a0, arow + kb, arow + kb + 8);
    ld_frag(a1, arow + kb + 32, arow + kb + 40);
#pragma unroll
    for (int n = 0; n < 4; ++n) {
      const u16* wr = wrow0 + (u32)(n * 16) * EMB + kb;
      ld_frag(b0[n], wr, wr + 8);
      ld_frag(b1[n], wr + 32, wr + 40);
    }
#pragma unroll
    for (int n = 0; n < 4; ++n)
      acc[n] = __builtin_amdgcn_wmma_f32_16x16x32_bf16(
          false, a0.bf, false, b0[n].bf, (short)0, acc[n], false, false);
#pragma unroll
    for (int n = 0; n < 4; ++n)
      acc[n] = __builtin_amdgcn_wmma_f32_16x16x32_bf16(
          false, a1.bf, false, b1[n].bf, (short)0, acc[n], false, false);
  }
#pragma unroll
  for (int n = 0; n < 4; ++n) {
    const int ng = nt * 64 + n * 16 + l16;
    const float bv = bias[ng];
    const int which = ng >> 10;                    // 0=q 1=k 2=v (uniform per tile)
    const int rem = ng & 1023;
    const int h = rem >> 6;
    const int d = rem & 63;
#pragma unroll
    for (int r = 0; r < 8; ++r) {                  // C/D: M = r + hi8, N = l16
      const int m = mt * 16 + r + hi8;
      const int b = m >> 11;
      const int s = m & 2047;
      const u16 val = f2bf(acc[n][r] + bv);
      const int bh = b * NH + h;
      if (which == 2) {                            // V transposed: [B,H,D,S]
        vo[(u32)(bh * HD + d) * S_LEN + s] = val;
      } else {
        u16* dst = (which == 0) ? qo : ko;         // Q,K: [B,H,S,D]
        dst[(u32)(bh * S_LEN + s) * HD + d] = val;
      }
    }
  }
}

// --------------------------- fused attention -------------------------------
// One wave per (b,h,16-query block). Per 32-key chunk:
//   scores = Q[16x64] @ K^T          (two independent 2-WMMA chains)
//   sample = (u < sigmoid(s/8)) via fma(u,e^-z,u) < 1  (bf16 0/1 tile in LDS)
//   out   += sample[16x32] @ V[32x64] (4 WMMA, contiguous B-frags from vT)
__global__ __launch_bounds__(128) void attn_kernel(
    const u16* __restrict__ qb_, const u16* __restrict__ kb_,
    const u16* __restrict__ vb_, u16* __restrict__ ao) {
  __shared__ __attribute__((aligned(16))) u16 samp[4][16 * 40];  // stride 40 (80B)
  const int lane = threadIdx.x & 31;
  const int wave = threadIdx.x >> 5;
  const int wid  = blockIdx.x * 4 + wave;          // 4096 waves
  const int qt   = wid & 127;
  const int h    = (wid >> 7) & 15;
  const int b    = wid >> 11;
  const int bh   = b * NH + h;
  const int l16  = lane & 15;
  const int hi8  = (lane >> 4) * 8;
  const int hi16 = hi8 * 2;

  // Q A-fragments for d 0..31 and 32..63 (contiguous 16B runs per lane)
  const u16* qrow = qb_ + (u32)(bh * S_LEN + qt * 16 + l16) * HD + hi16;
  FragBF qa0, qa1;
  ld_frag(qa0, qrow, qrow + 8);
  ld_frag(qa1, qrow + 32, qrow + 40);

  v8f acc[4] = {zero_v8f(), zero_v8f(), zero_v8f(), zero_v8f()};
  const u16* kbase  = kb_ + (u32)bh * S_LEN * HD;      // [S,D]
  const u16* vtbase = vb_ + (u32)bh * HD * S_LEN;      // [D,S] (transposed)
  const u16* krow_l = kbase + (u32)l16 * HD + hi16;    // per-lane K row base
  u16* srow_base = &samp[wave][0];
  const u32 idx_base = ((u32)bh * 2048u + (u32)(qt * 16 + hi8)) * 2048u;

  for (int kc = 0; kc < S_LEN; kc += 32) {
    // prefetch next chunk's K rows / V columns into cache (global_prefetch_b8)
    if (kc + 32 < S_LEN) {
      __builtin_prefetch(kbase + (u32)(kc + 32 + l16) * HD, 0, 1);
      __builtin_prefetch(vtbase + (u32)(l16 * 2) * S_LEN + kc + 32, 0, 1);
    }
    // K B-fragments for both 16-key tiles, both d-halves (8 b128 loads)
    const u16* kr0 = krow_l + (u32)kc * HD;            // key tile 0
    const u16* kr1 = kr0 + 16 * HD;                    // key tile 1
    FragBF kf00, kf01, kf10, kf11;
    ld_frag(kf00, kr0, kr0 + 8);
    ld_frag(kf01, kr0 + 32, kr0 + 40);
    ld_frag(kf10, kr1, kr1 + 8);
    ld_frag(kf11, kr1 + 32, kr1 + 40);
    v8f sc0 = zero_v8f(), sc1 = zero_v8f();            // independent chains
    sc0 = __builtin_amdgcn_wmma_f32_16x16x32_bf16(false, qa0.bf, false, kf00.bf,
                                                  (short)0, sc0, false, false);
    sc1 = __builtin_amdgcn_wmma_f32_16x16x32_bf16(false, qa0.bf, false, kf10.bf,
                                                  (short)0, sc1, false, false);
    sc0 = __builtin_amdgcn_wmma_f32_16x16x32_bf16(false, qa1.bf, false, kf01.bf,
                                                  (short)0, sc0, false, false);
    sc1 = __builtin_amdgcn_wmma_f32_16x16x32_bf16(false, qa1.bf, false, kf11.bf,
                                                  (short)0, sc1, false, false);
    // sigmoid + Bernoulli: u < 1/(1+e)  <=>  u + u*e < 1
#pragma unroll
    for (int r = 0; r < 8; ++r) {                  // C/D: M=r+hi8, N=l16
      const u32 idx0 = idx_base + (u32)(r * 2048) + (u32)(kc + l16);
      const float e0 = __expf(sc0[r] * -0.125f);
      const float e1 = __expf(sc1[r] * -0.125f);
      const float u0 = (float)pcg(idx0 * 0x9E3779B9u + 0x6A09E667u) *
                       2.3283064365386963e-10f;    // 2^-32
      const float u1 = (float)pcg((idx0 + 16u) * 0x9E3779B9u + 0x6A09E667u) *
                       2.3283064365386963e-10f;
      const u16 s0 = (__fmaf_rn(u0, e0, u0) < 1.0f) ? (u16)0x3F80 : (u16)0;
      const u16 s1 = (__fmaf_rn(u1, e1, u1) < 1.0f) ? (u16)0x3F80 : (u16)0;
      srow_base[(r + hi8) * 40 + l16] = s0;
      srow_base[(r + hi8) * 40 + 16 + l16] = s1;
    }
    // cross-lane LDS round-trip: DS ops are in-order per wave; drain stores
    asm volatile("s_wait_dscnt 0x0" ::: "memory");
    FragBF sa;                                     // sample in A-layout
    {
      const u16* srow = srow_base + l16 * 40 + hi16;
      ld_frag(sa, srow, srow + 8);
    }
    // V B-fragments from vT: lane row d = n*16+l16, keys contiguous along S
    FragBF vf[4];
#pragma unroll
    for (int n = 0; n < 4; ++n) {
      const u16* vtrow = vtbase + (u32)(n * 16 + l16) * S_LEN + kc + hi16;
      ld_frag(vf[n], vtrow, vtrow + 8);
    }
#pragma unroll
    for (int n = 0; n < 4; ++n)
      acc[n] = __builtin_amdgcn_wmma_f32_16x16x32_bf16(
          false, sa.bf, false, vf[n].bf, (short)0, acc[n], false, false);
  }

  // store out[b,h,q,d] -> attn_bf16[(b*S+q)*E + h*64 + d]
#pragma unroll
  for (int n = 0; n < 4; ++n) {
    const int d = h * HD + n * 16 + l16;
#pragma unroll
    for (int r = 0; r < 8; ++r) {
      const int s = qt * 16 + r + hi8;
      ao[(u32)(b * S_LEN + s) * EMB + d] = f2bf(acc[n][r]);
    }
  }
}

// ------------------------------ output GEMM --------------------------------
__global__ __launch_bounds__(128) void out_gemm_kernel(
    const u16* __restrict__ ab, const u16* __restrict__ wb,
    const float* __restrict__ bias, float* __restrict__ out) {
  const int lane = threadIdx.x & 31;
  const int wave = threadIdx.x >> 5;
  const int wid  = blockIdx.x * 4 + wave;          // 4096 waves
  const int nt   = wid & 15;                       // 16 tiles of N=64
  const int mt   = wid >> 4;                       // 256 tiles of M=16
  const int l16  = lane & 15;
  const int hi8  = (lane >> 4) * 8;
  const int hi16 = hi8 * 2;

  const u16* arow = ab + (u32)(mt * 16 + l16) * EMB + hi16;
  const u16* wrow0 = wb + (u32)(nt * 64 + l16) * EMB + hi16;
  v8f acc[4] = {zero_v8f(), zero_v8f(), zero_v8f(), zero_v8f()};

  for (int kb = 0; kb < EMB; kb += 64) {
    FragBF a0, a1, b0[4], b1[4];
    ld_frag(a0, arow + kb, arow + kb + 8);
    ld_frag(a1, arow + kb + 32, arow + kb + 40);
#pragma unroll
    for (int n = 0; n < 4; ++n) {
      const u16* wr = wrow0 + (u32)(n * 16) * EMB + kb;
      ld_frag(b0[n], wr, wr + 8);
      ld_frag(b1[n], wr + 32, wr + 40);
    }
#pragma unroll
    for (int n = 0; n < 4; ++n)
      acc[n] = __builtin_amdgcn_wmma_f32_16x16x32_bf16(
          false, a0.bf, false, b0[n].bf, (short)0, acc[n], false, false);
#pragma unroll
    for (int n = 0; n < 4; ++n)
      acc[n] = __builtin_amdgcn_wmma_f32_16x16x32_bf16(
          false, a1.bf, false, b1[n].bf, (short)0, acc[n], false, false);
  }
#pragma unroll
  for (int n = 0; n < 4; ++n) {
    const int ng = nt * 64 + n * 16 + l16;
    const float bv = bias[ng];
#pragma unroll
    for (int r = 0; r < 8; ++r) {
      const int m = mt * 16 + r + hi8;
      out[(u32)m * EMB + ng] = acc[n][r] + bv;
    }
  }
}

// ------------------------------- launcher ----------------------------------
extern "C" void kernel_launch(void* const* d_in, const int* in_sizes, int n_in,
                              void* d_out, int out_size, void* d_ws,
                              size_t ws_size, hipStream_t stream) {
  (void)in_sizes; (void)n_in; (void)out_size; (void)ws_size;
  const float* x      = (const float*)d_in[0];
  const float* qkv_w  = (const float*)d_in[3];
  const float* qkv_b  = (const float*)d_in[4];
  const float* out_w  = (const float*)d_in[5];
  const float* out_b  = (const float*)d_in[6];

  char* ws = (char*)d_ws;
  u16* xb   = (u16*)(ws + 0);           //  8 MB  x bf16
  u16* wqkv = (u16*)(ws + 8388608);     //  6 MB  qkv_w bf16
  u16* wout = (u16*)(ws + 14680064);    //  2 MB  out_w bf16
  u16* qb   = (u16*)(ws + 16777216);    //  8 MB  q [B,H,S,D]
  u16* kb   = (u16*)(ws + 25165824);    //  8 MB  k [B,H,S,D]
  u16* vb   = (u16*)(ws + 33554432);    //  8 MB  v [B,H,D,S] (transposed)
  u16* ab   = (u16*)(ws + 41943040);    //  8 MB  attn out bf16 [B*S, E]

  cvt_bf16_kernel<<<4096, 256, 0, stream>>>(x, xb, (BATCH * S_LEN * EMB) / 4);
  cvt_bf16_kernel<<<3072, 256, 0, stream>>>(qkv_w, wqkv, (3 * EMB * EMB) / 4);
  cvt_bf16_kernel<<<1024, 256, 0, stream>>>(out_w, wout, (EMB * EMB) / 4);

  qkv_gemm_kernel<<<3072, 128, 0, stream>>>(xb, wqkv, qkv_b, qb, kb, vb);
  attn_kernel<<<1024, 128, 0, stream>>>(qb, kb, vb, ab);
  out_gemm_kernel<<<1024, 128, 0, stream>>>(ab, wout, out_b, (float*)d_out);
}
